// SimpleLinearRecurrence_57552561766978
// MI455X (gfx1250) — compile-verified, matrix operand order checked by reading, not run on hardware
//
#include <hip/hip_runtime.h>
#include <hip/hip_bf16.h>

typedef __bf16 bf16;
typedef __attribute__((ext_vector_type(16))) __bf16 v16bf;
typedef __attribute__((ext_vector_type(8)))  __bf16 v8bf;
typedef __attribute__((ext_vector_type(8)))  float  v8f;
typedef __attribute__((ext_vector_type(4)))  int    v4i;

#define DIMN   1024
#define BATCH  4
#define SEQ    4096
#define MTOT   (BATCH*SEQ)      // 16384 rows
#define NCHUNK 16
#define CLEN   (SEQ/NCHUNK)     // 256

// LDS B-tile: 64 rows x 32 k, row stride 40 bf16 (80B) -> conflict-free b128 reads
#define BS_STRIDE 40

#if __has_builtin(__builtin_amdgcn_global_load_async_to_lds_b128)
#define USE_ASYNC_LDS 1
typedef __attribute__((address_space(1))) v4i as1_v4i;
typedef __attribute__((address_space(3))) v4i as3_v4i;
#else
#define USE_ASYNC_LDS 0
#endif

// ---------------- f32 -> bf16 conversion ----------------
__global__ void cvt_f32_bf16(const float* __restrict__ in, bf16* __restrict__ out, int n) {
    int i = blockIdx.x * blockDim.x + threadIdx.x;
    if (i < n) out[i] = (bf16)in[i];
}

// ---------------- WMMA fragment loads --------------------
// 16-bit A/B fragment (16x32 tile, K contiguous): lane l=lane%16 holds row l;
// the lane's 16 values are two contiguous 8-elem chunks at K offsets h*8 and
// h*8+16 (h = lane/16), per the ISA 16-bit A-matrix VGPR layout.
__device__ __forceinline__ v16bf load_frag_g(const bf16* __restrict__ p) {
    v8bf lo = *(const v8bf*)p;
    v8bf hi = *(const v8bf*)(p + 16);
    return __builtin_shufflevector(lo, hi, 0,1,2,3,4,5,6,7,8,9,10,11,12,13,14,15);
}
__device__ __forceinline__ v16bf load_frag_lds(const bf16* p) {
    v8bf lo = *(const v8bf*)p;           // ds_load_b128
    v8bf hi = *(const v8bf*)(p + 16);    // ds_load_b128
    return __builtin_shufflevector(lo, hi, 0,1,2,3,4,5,6,7,8,9,10,11,12,13,14,15);
}

// ---------------- TN GEMM: C[M,N] = A[M,K] * Bw[N,K]^T ----------------
// EPI: 0 -> bf16 store, 1 -> sigmoid then bf16 store, 2 -> f32 store
// Block: 256 threads = 8 waves; wave = 16(M) x 64(N) strip; block = 128x64.
// B tile (64x32) staged in double-buffered LDS via async loads (ASYNCcnt).
template<int EPI>
__global__ __launch_bounds__(256)
void gemm_bf16_wmma(const bf16* __restrict__ A, const bf16* __restrict__ Bw,
                    void* __restrict__ outv, int M, int N, int K) {
    __shared__ bf16 Bs[2][64 * BS_STRIDE];

    const int lane = threadIdx.x & 31;
    const int wave = threadIdx.x >> 5;
    const int l = lane & 15;
    const int h = lane >> 4;
    const int m0 = blockIdx.y * 128 + wave * 16;
    const int n0 = blockIdx.x * 64;

    // --- staging addresses: each of the 256 threads moves one 16B chunk/step
    const int srow = threadIdx.x >> 2;            // 0..63  (B row within tile)
    const int sseg = threadIdx.x & 3;             // 0..3   (16B segment in 32-k row)
    const bf16* gstage = Bw + (size_t)(n0 + srow) * K + sseg * 8;
    bf16* lstage0 = &Bs[0][srow * BS_STRIDE + sseg * 8];
    bf16* lstage1 = &Bs[1][srow * BS_STRIDE + sseg * 8];

    // --- per-wave A pointer (direct global b128 loads)
    const bf16* ap = A + (size_t)(m0 + l) * K + h * 8;

    auto stage = [&](int buf, int kb) {
        const bf16* gp = gstage + kb;
        bf16* lp = buf ? lstage1 : lstage0;
#if USE_ASYNC_LDS
        __builtin_amdgcn_global_load_async_to_lds_b128((as1_v4i*)gp, (as3_v4i*)lp, 0, 0);
#else
        *(v8bf*)lp = *(const v8bf*)gp;
#endif
    };
    auto wait_stage = [&]() {
#if USE_ASYNC_LDS
#if __has_builtin(__builtin_amdgcn_s_wait_asynccnt)
        __builtin_amdgcn_s_wait_asynccnt(0);
#else
        asm volatile("s_wait_asynccnt 0x0" ::: "memory");
#endif
#endif
        __syncthreads();
    };

    v8f acc[4] = {v8f{}, v8f{}, v8f{}, v8f{}};

    stage(0, 0);
    for (int kb = 0; kb < K; kb += 32) {
        const int p = (kb >> 5) & 1;
        wait_stage();                      // staged tile visible to all waves
        if (kb + 32 < K) {
            stage(p ^ 1, kb + 32);         // overlap next-tile DMA with WMMAs
            __builtin_prefetch(ap + kb + 32, 0, 3);
        }

        v16bf a = load_frag_g(ap + kb);
        #pragma unroll
        for (int j = 0; j < 4; ++j) {
            const bf16* bs = &Bs[p][(j * 16 + l) * BS_STRIDE + h * 8];
            v16bf b = load_frag_lds(bs);
            acc[j] = __builtin_amdgcn_wmma_f32_16x16x32_bf16(
                         false, a, false, b, (short)0, acc[j], false, false);
        }
        __syncthreads();                   // all reads done before buffer reuse
    }

    // C layout: VGPR i -> row m0 + i + 8*h, col n0 + j*16 + l
    #pragma unroll
    for (int j = 0; j < 4; ++j) {
        #pragma unroll
        for (int i = 0; i < 8; ++i) {
            const int m = m0 + i + h * 8;
            const int n = n0 + j * 16 + l;
            float val = acc[j][i];
            if (EPI == 1) val = 1.0f / (1.0f + __expf(-val));
            if (EPI == 2) {
                ((float*)outv)[(size_t)m * N + n] = val;
            } else {
                ((bf16*)outv)[(size_t)m * N + n] = (bf16)val;
            }
        }
    }
}

// ---------------- chunked linear scan ----------------
// state_{t+1} = decay_d * state_t + k_t*v_t ;  wkv_t = state_t + eu_d * k_t*v_t
__global__ void scan_pass1(const bf16* __restrict__ kb_, const bf16* __restrict__ vb_,
                           const float* __restrict__ td, float* __restrict__ carr) {
    int idx = blockIdx.x * blockDim.x + threadIdx.x;       // (b, chunk, d), d fastest
    if (idx >= BATCH * NCHUNK * DIMN) return;
    const int d  = idx & (DIMN - 1);
    const int bc = idx / DIMN;
    const int chunk = bc & (NCHUNK - 1);
    const int b = bc >> 4;
    const float dec = __expf(-__expf(td[d]));
    size_t base = ((size_t)b * SEQ + (size_t)chunk * CLEN) * DIMN + d;
    float s = 0.f;
    for (int t = 0; t < CLEN; ++t) {
        float kv = (float)kb_[base] * (float)vb_[base];
        s = dec * s + kv;
        base += DIMN;
    }
    carr[idx] = s;
}

__global__ void scan_pass2(const float* __restrict__ td, float* __restrict__ carr) {
    int idx = blockIdx.x * blockDim.x + threadIdx.x;       // b*DIMN + d
    if (idx >= BATCH * DIMN) return;
    const int d = idx & (DIMN - 1);
    const int b = idx >> 10;
    const float w = -__expf(td[d]);
    const float decL = __expf(w * (float)CLEN);            // dec^CLEN
    float c = 0.f;
    for (int chunk = 0; chunk < NCHUNK; ++chunk) {
        size_t off = ((size_t)b * NCHUNK + chunk) * DIMN + d;
        float e = carr[off];
        carr[off] = c;
        c = decL * c + e;
    }
}

__global__ void scan_pass3(const bf16* __restrict__ kb_, const bf16* __restrict__ vb_,
                           const bf16* __restrict__ rb_,
                           const float* __restrict__ td, const float* __restrict__ tf,
                           const float* __restrict__ carr, bf16* __restrict__ op) {
    int idx = blockIdx.x * blockDim.x + threadIdx.x;       // (b, chunk, d)
    if (idx >= BATCH * NCHUNK * DIMN) return;
    const int d  = idx & (DIMN - 1);
    const int bc = idx / DIMN;
    const int chunk = bc & (NCHUNK - 1);
    const int b = bc >> 4;
    const float dec = __expf(-__expf(td[d]));
    const float eu  = __expf(tf[d]);
    size_t base = ((size_t)b * SEQ + (size_t)chunk * CLEN) * DIMN + d;
    float s = carr[idx];
    for (int t = 0; t < CLEN; ++t) {
        float kv  = (float)kb_[base] * (float)vb_[base];
        float wkv = s + eu * kv;
        s = dec * s + kv;
        op[base] = (bf16)((float)rb_[base] * wkv);
        base += DIMN;
    }
}

// ---------------- launcher ----------------
extern "C" void kernel_launch(void* const* d_in, const int* in_sizes, int n_in,
                              void* d_out, int out_size, void* d_ws, size_t ws_size,
                              hipStream_t stream) {
    const float* x  = (const float*)d_in[0];
    const float* Wk = (const float*)d_in[1];
    const float* Wv = (const float*)d_in[2];
    const float* Wr = (const float*)d_in[3];
    const float* Wo = (const float*)d_in[4];
    const float* td = (const float*)d_in[5];
    const float* tf = (const float*)d_in[6];

    const size_t MK = (size_t)MTOT * DIMN;   // 16.7M
    const size_t DD = (size_t)DIMN * DIMN;   // 1.05M

    char* p = (char*)d_ws;
    bf16* xb  = (bf16*)p; p += MK * sizeof(bf16);
    bf16* wkb = (bf16*)p; p += DD * sizeof(bf16);
    bf16* wvb = (bf16*)p; p += DD * sizeof(bf16);
    bf16* wrb = (bf16*)p; p += DD * sizeof(bf16);
    bf16* wob = (bf16*)p; p += DD * sizeof(bf16);
    bf16* kb  = (bf16*)p; p += MK * sizeof(bf16);
    bf16* vb  = (bf16*)p; p += MK * sizeof(bf16);
    bf16* rb  = (bf16*)p; p += MK * sizeof(bf16);
    bf16* opb = (bf16*)p; p += MK * sizeof(bf16);
    float* carr = (float*)p;                 // BATCH*NCHUNK*DIMN floats

    // 1) convert inputs to bf16
    cvt_f32_bf16<<<(int)((MK + 255) / 256), 256, 0, stream>>>(x,  xb,  (int)MK);
    cvt_f32_bf16<<<(int)((DD + 255) / 256), 256, 0, stream>>>(Wk, wkb, (int)DD);
    cvt_f32_bf16<<<(int)((DD + 255) / 256), 256, 0, stream>>>(Wv, wvb, (int)DD);
    cvt_f32_bf16<<<(int)((DD + 255) / 256), 256, 0, stream>>>(Wr, wrb, (int)DD);
    cvt_f32_bf16<<<(int)((DD + 255) / 256), 256, 0, stream>>>(Wo, wob, (int)DD);

    // 2) projections via WMMA
    dim3 g(DIMN / 64, MTOT / 128);
    gemm_bf16_wmma<0><<<g, 256, 0, stream>>>(xb, wkb, kb, MTOT, DIMN, DIMN);
    gemm_bf16_wmma<0><<<g, 256, 0, stream>>>(xb, wvb, vb, MTOT, DIMN, DIMN);
    gemm_bf16_wmma<1><<<g, 256, 0, stream>>>(xb, wrb, rb, MTOT, DIMN, DIMN);

    // 3) chunked scan
    scan_pass1<<<(BATCH * NCHUNK * DIMN) / 256, 256, 0, stream>>>(kb, vb, td, carr);
    scan_pass2<<<(BATCH * DIMN + 255) / 256, 256, 0, stream>>>(td, carr);
    scan_pass3<<<(BATCH * NCHUNK * DIMN) / 256, 256, 0, stream>>>(kb, vb, rb, td, tf, carr, opb);

    // 4) output projection -> d_out (f32)
    gemm_bf16_wmma<2><<<g, 256, 0, stream>>>(opb, wob, d_out, MTOT, DIMN, DIMN);
}
